// GRITAttention_21088289423720
// MI455X (gfx1250) — matrix operand order repaired and checked by reading, not compiled, not run on hardware
//
#include <hip/hip_runtime.h>
#include <math.h>
#include <stdint.h>

// ---------------------------------------------------------------------------
// GRIT attention for MI455X (gfx1250, wave32, WMMA bf16 16x16x32,
// async global->LDS staging with ASYNCcnt double buffering)
// N=4096, D=256, DPE=16, H=8, HD=32, SCALE=1/sqrt(32)
// ---------------------------------------------------------------------------

#define NN 4096
#define DD 256
#define HH 8
#define HD 32
#define D3 768
#define SCALE 0.17677669529663688f

typedef __bf16 bf16;
typedef __attribute__((ext_vector_type(16))) __bf16 v16bf;
typedef __attribute__((ext_vector_type(8)))  float  v8f;
typedef int v4i __attribute__((vector_size(16)));

// 32-byte WMMA operand: 16 bf16 halves, loadable as two uint4
union AV {
    v16bf v;
    uint4 u[2];
};

__device__ __forceinline__ v8f wmma_bf16(const AV& a, const AV& b, v8f c) {
    return __builtin_amdgcn_wmma_f32_16x16x32_bf16(
        /*neg_a=*/false, a.v, /*neg_b=*/false, b.v,
        /*c_mod=*/(short)0, c, /*reuse_a=*/false, /*reuse_b=*/false);
}

// ---- gfx1250 async global->LDS copy (16B per lane), ASYNCcnt-tracked ------
__device__ __forceinline__ void async_copy_b128(const void* gsrc, void* ldst) {
#if __has_builtin(__builtin_amdgcn_global_load_async_to_lds_b128)
    __builtin_amdgcn_global_load_async_to_lds_b128(
        (__attribute__((address_space(1))) v4i*)(uintptr_t)gsrc,
        (__attribute__((address_space(3))) v4i*)(uint32_t)(uintptr_t)ldst,
        0, 0);
#else
    // Fallback: synchronous load + ds_store (correct, just not async)
    *(uint4*)ldst = *(const uint4*)gsrc;
#endif
}

__device__ __forceinline__ void wait_async_le(int n) {
#if __has_builtin(__builtin_amdgcn_s_wait_asynccnt)
    if (n == 0) __builtin_amdgcn_s_wait_asynccnt(0);
    else        __builtin_amdgcn_s_wait_asynccnt(1);
#else
    if (n == 0) asm volatile("s_wait_asynccnt 0x0" ::: "memory");
    else        asm volatile("s_wait_asynccnt 0x1" ::: "memory");
#endif
}

// ---------------------------------------------------------------------------
// Kernel 1: convert x -> bf16 row-major; Wqkv, Wout -> bf16 TRANSPOSED
// ---------------------------------------------------------------------------
__global__ __launch_bounds__(256) void prep_convert(
    const float* __restrict__ x, const float* __restrict__ Wqkv,
    const float* __restrict__ Wout,
    bf16* __restrict__ xb, bf16* __restrict__ wqkvT, bf16* __restrict__ woutT) {
    int i = blockIdx.x * blockDim.x + threadIdx.x;             // up to 1M
    if (i < NN * DD) xb[i] = (bf16)x[i];
    if (i < D3 * DD) {                                          // wqkvT[n][k]
        int n = i >> 8, k = i & 255;
        wqkvT[i] = (bf16)Wqkv[(size_t)k * D3 + n];
    }
    if (i < DD * DD) {                                          // woutT[n][k]
        int n = i >> 8, k = i & 255;
        woutT[i] = (bf16)Wout[(size_t)k * DD + n];
    }
}

// ---------------------------------------------------------------------------
// Kernel 2: pe_proj[h][n] = pe[n,:] @ Wpe[:,h] + bpe[h]   (fp32, tiny)
// ---------------------------------------------------------------------------
__global__ __launch_bounds__(256) void pe_proj_kernel(
    const float* __restrict__ pe, const float* __restrict__ Wpe,
    const float* __restrict__ bpe, float* __restrict__ ppT) {
    int n = blockIdx.x * blockDim.x + threadIdx.x;
    if (n >= NN) return;
    float v[16];
#pragma unroll
    for (int t = 0; t < 16; ++t) v[t] = pe[n * 16 + t];
#pragma unroll
    for (int h = 0; h < HH; ++h) {
        float s = bpe[h];
#pragma unroll
        for (int t = 0; t < 16; ++t) s += v[t] * Wpe[t * HH + h];
        ppT[(size_t)h * NN + n] = s;
    }
}

// ---------------------------------------------------------------------------
// Kernel 3: QKV projection. One wave per 16x16 tile of [4096 x 768].
// Q (pre-scaled) / K as [H][N][HD] bf16, V transposed as [H][HD][N] bf16.
// ---------------------------------------------------------------------------
__global__ __launch_bounds__(256) void qkv_proj_kernel(
    const bf16* __restrict__ xb, const bf16* __restrict__ wqkvT,
    const float* __restrict__ bqkv,
    bf16* __restrict__ Qb, bf16* __restrict__ Kb, bf16* __restrict__ Vt) {
    const int lane = threadIdx.x & 31, wv = threadIdx.x >> 5;
    const int gw = blockIdx.x * 8 + wv;                 // 12288 waves
    const int mt = gw / (D3 / 16), nt = gw % (D3 / 16);
    const int m0 = mt * 16, n0 = nt * 16;
    const int hf = lane >> 4, lm = lane & 15, kbase = hf * 8;

    v8f c = {};
#pragma unroll
    for (int kk = 0; kk < DD; kk += 32) {
        AV a, b;
        const bf16* ar = xb + (size_t)(m0 + lm) * DD + kk;
        a.u[0] = *(const uint4*)(ar + 2 * kbase);
        a.u[1] = *(const uint4*)(ar + 2 * kbase + 8);
        const bf16* br = wqkvT + (size_t)(n0 + lm) * DD + kk + hf * 16;
        b.u[0] = *(const uint4*)(br);
        b.u[1] = *(const uint4*)(br + 8);
        c = wmma_bf16(a, b, c);
    }
    const int n = n0 + lm;
    const float bias = bqkv[n];
    const int part = n >> 8, h = (n >> 5) & 7, d = n & 31;
#pragma unroll
    for (int r = 0; r < 8; ++r) {
        int row = m0 + r + 8 * hf;
        float val = c[r] + bias;
        if (part == 0)
            Qb[((size_t)h * NN + row) * HD + d] = (bf16)(val * SCALE);
        else if (part == 1)
            Kb[((size_t)h * NN + row) * HD + d] = (bf16)val;
        else
            Vt[((size_t)h * HD + d) * NN + row] = (bf16)val;
    }
}

// ---------------------------------------------------------------------------
// Kernel 4: flash attention. 8 waves/block, all on the same head; K/V chunks
// are staged cooperatively into LDS with async global->LDS b128 copies,
// double-buffered on ASYNCcnt. Per chunk per wave: 2x WMMA QK^T, online
// softmax, LDS relayout P -> A-layout, 2x WMMA PV.
// ---------------------------------------------------------------------------
__global__ __launch_bounds__(256) void attn_kernel(
    const bf16* __restrict__ Qb, const bf16* __restrict__ Kb,
    const bf16* __restrict__ Vt, const float* __restrict__ ppT,
    bf16* __restrict__ AO) {
    __shared__ bf16 Kbuf[2][32][32];                    // 4 KB (j-local x d)
    __shared__ bf16 Vbuf[2][32][32];                    // 4 KB (d x j-local)
    __shared__ bf16 ldsP[8][16][32];                    // 8 KB, 1 KB per wave

    const int tid = threadIdx.x;
    const int lane = tid & 31, wv = tid >> 5;
    const int gw = blockIdx.x * 8 + wv;                 // 2048 waves
    const int h = gw >> 8;                              // same for whole block
    const int i0 = (gw & 255) << 4;
    const int hf = lane >> 4, lm = lane & 15, kbase = hf * 8;

    const bf16* kh = Kb + (size_t)h * NN * HD;
    const bf16* vh = Vt + (size_t)h * HD * NN;
    const float* pph = ppT + (size_t)h * NN;

    // staging role of this thread: threads 0..127 stage K, 128..255 stage V
    const int srow = (tid & 127) >> 2;                  // 0..31
    const int sseg = (tid & 3) * 8;                     // 0,8,16,24 (bf16 elems)
    const bool stageK = tid < 128;

    // Q tile in A layout (SCALE already folded in)
    AV aq;
    const bf16* qrow = Qb + ((size_t)h * NN + i0 + lm) * HD;
    aq.u[0] = *(const uint4*)(qrow + 2 * kbase);
    aq.u[1] = *(const uint4*)(qrow + 2 * kbase + 8);

    float pi[8];
#pragma unroll
    for (int r = 0; r < 8; ++r) pi[r] = pph[i0 + r + 8 * hf];

    v8f o0 = {}, o1 = {};
    float m[8], l[8];
#pragma unroll
    for (int r = 0; r < 8; ++r) { m[r] = -1e30f; l[r] = 0.0f; }

    // prologue: stage chunk 0 into buffer 0
    if (stageK)
        async_copy_b128(kh + (size_t)srow * HD + sseg, &Kbuf[0][srow][sseg]);
    else
        async_copy_b128(vh + (size_t)srow * NN + sseg, &Vbuf[0][srow][sseg]);

    for (int jc = 0; jc < NN; jc += 32) {
        const int cur = (jc >> 5) & 1;
        // stage next chunk into the other buffer, then wait for current
        if (jc + 32 < NN) {
            const int nxt = cur ^ 1;
            if (stageK)
                async_copy_b128(kh + (size_t)(jc + 32 + srow) * HD + sseg,
                                &Kbuf[nxt][srow][sseg]);
            else
                async_copy_b128(vh + (size_t)srow * NN + (jc + 32) + sseg,
                                &Vbuf[nxt][srow][sseg]);
            wait_async_le(1);
        } else {
            wait_async_le(0);
        }
        __syncthreads();   // chunk `cur` fully in LDS for all waves

        // K^T B-tiles from LDS: lane = column j, halves = d (32B contiguous)
        AV bk0, bk1;
        {
            const bf16* kr0 = &Kbuf[cur][lm][hf * 16];
            const bf16* kr1 = &Kbuf[cur][16 + lm][hf * 16];
            bk0.u[0] = *(const uint4*)(kr0); bk0.u[1] = *(const uint4*)(kr0 + 8);
            bk1.u[0] = *(const uint4*)(kr1); bk1.u[1] = *(const uint4*)(kr1 + 8);
        }

        v8f z = {};
        v8f s0 = wmma_bf16(aq, bk0, z);
        v8f s1 = wmma_bf16(aq, bk1, z);

        const float pj0 = pph[jc + lm];
        const float pj1 = pph[jc + 16 + lm];

#pragma unroll
        for (int r = 0; r < 8; ++r) {
            float e0 = s0[r] + pi[r] - pj0;
            float e1 = s1[r] + pi[r] - pj1;
            // row max across the 16 lanes of this half-wave (masks < 16)
            float mc = fmaxf(e0, e1);
            mc = fmaxf(mc, __shfl_xor(mc, 1, 32));
            mc = fmaxf(mc, __shfl_xor(mc, 2, 32));
            mc = fmaxf(mc, __shfl_xor(mc, 4, 32));
            mc = fmaxf(mc, __shfl_xor(mc, 8, 32));
            float mnew = fmaxf(m[r], mc);
            float corr = __expf(m[r] - mnew);
            float p0 = __expf(e0 - mnew);
            float p1 = __expf(e1 - mnew);
            float sum = p0 + p1;
            sum += __shfl_xor(sum, 1, 32);
            sum += __shfl_xor(sum, 2, 32);
            sum += __shfl_xor(sum, 4, 32);
            sum += __shfl_xor(sum, 8, 32);
            m[r] = mnew;
            l[r] = l[r] * corr + sum;
            o0[r] *= corr;
            o1[r] *= corr;
            int row = r + 8 * hf;
            ldsP[wv][row][lm]      = (bf16)p0;
            ldsP[wv][row][lm + 16] = (bf16)p1;
        }

        // Re-read P in A layout (per-wave LDS round trip)
        AV ap;
        {
            const bf16* pr = &ldsP[wv][lm][0];
            ap.u[0] = *(const uint4*)(pr + 2 * kbase);
            ap.u[1] = *(const uint4*)(pr + 2 * kbase + 8);
        }

        // V B-tiles from LDS: lane = column d, halves = j (32B contiguous)
        AV bv0, bv1;
        {
            const bf16* vr0 = &Vbuf[cur][lm][hf * 16];
            const bf16* vr1 = &Vbuf[cur][16 + lm][hf * 16];
            bv0.u[0] = *(const uint4*)(vr0); bv0.u[1] = *(const uint4*)(vr0 + 8);
            bv1.u[0] = *(const uint4*)(vr1); bv1.u[1] = *(const uint4*)(vr1 + 8);
        }

        o0 = wmma_bf16(ap, bv0, o0);
        o1 = wmma_bf16(ap, bv1, o1);

        __syncthreads();   // everyone done reading buf `cur` before re-stage
    }

#pragma unroll
    for (int r = 0; r < 8; ++r) {
        float inv = 1.0f / l[r];
        int row = i0 + r + 8 * hf;
        AO[(size_t)row * DD + h * HD + lm]      = (bf16)(o0[r] * inv);
        AO[(size_t)row * DD + h * HD + 16 + lm] = (bf16)(o1[r] * inv);
    }
}

// ---------------------------------------------------------------------------
// Kernel 5: output projection [4096x256] @ [256x256] + bout -> fp32 d_out
// ---------------------------------------------------------------------------
__global__ __launch_bounds__(256) void out_proj_kernel(
    const bf16* __restrict__ AO, const bf16* __restrict__ woutT,
    const float* __restrict__ bout, float* __restrict__ out) {
    const int lane = threadIdx.x & 31, wv = threadIdx.x >> 5;
    const int gw = blockIdx.x * 8 + wv;                 // 4096 waves
    const int mt = gw >> 4, nt = gw & 15;
    const int m0 = mt * 16, n0 = nt * 16;
    const int hf = lane >> 4, lm = lane & 15, kbase = hf * 8;

    v8f c = {};
#pragma unroll
    for (int kk = 0; kk < DD; kk += 32) {
        AV a, b;
        const bf16* ar = AO + (size_t)(m0 + lm) * DD + kk;
        a.u[0] = *(const uint4*)(ar + 2 * kbase);
        a.u[1] = *(const uint4*)(ar + 2 * kbase + 8);
        const bf16* br = woutT + (size_t)(n0 + lm) * DD + kk + hf * 16;
        b.u[0] = *(const uint4*)(br);
        b.u[1] = *(const uint4*)(br + 8);
        c = wmma_bf16(a, b, c);
    }
    const int n = n0 + lm;
    const float bias = bout[n];
#pragma unroll
    for (int r = 0; r < 8; ++r) {
        int row = m0 + r + 8 * hf;
        out[(size_t)row * DD + n] = c[r] + bias;
    }
}

// ---------------------------------------------------------------------------
// Launch
// ---------------------------------------------------------------------------
extern "C" void kernel_launch(void* const* d_in, const int* in_sizes, int n_in,
                              void* d_out, int out_size, void* d_ws, size_t ws_size,
                              hipStream_t stream) {
    const float* x    = (const float*)d_in[0];
    const float* pe   = (const float*)d_in[1];
    const float* Wqkv = (const float*)d_in[2];
    const float* bqkv = (const float*)d_in[3];
    const float* Wpe  = (const float*)d_in[4];
    const float* bpe  = (const float*)d_in[5];
    const float* Wout = (const float*)d_in[6];
    const float* bout = (const float*)d_in[7];
    float* out = (float*)d_out;

    char* ws = (char*)d_ws;
    size_t off = 0;
    bf16* xb    = (bf16*)(ws + off); off += (size_t)NN * DD * 2;      // 2 MB
    bf16* wqkvT = (bf16*)(ws + off); off += (size_t)D3 * DD * 2;      // 384 KB
    bf16* woutT = (bf16*)(ws + off); off += (size_t)DD * DD * 2;      // 128 KB
    bf16* Qb    = (bf16*)(ws + off); off += (size_t)HH * NN * HD * 2; // 2 MB
    bf16* Kb    = (bf16*)(ws + off); off += (size_t)HH * NN * HD * 2; // 2 MB
    bf16* Vt    = (bf16*)(ws + off); off += (size_t)HH * NN * HD * 2; // 2 MB
    float* ppT  = (float*)(ws + off); off += (size_t)HH * NN * 4;     // 128 KB
    bf16* AO    = xb;  // x_bf16 is dead after qkv_proj; reuse its 2 MB

    prep_convert<<<(NN * DD + 255) / 256, 256, 0, stream>>>(x, Wqkv, Wout,
                                                            xb, wqkvT, woutT);
    pe_proj_kernel<<<NN / 256, 256, 0, stream>>>(pe, Wpe, bpe, ppT);
    qkv_proj_kernel<<<12288 / 8, 256, 0, stream>>>(xb, wqkvT, bqkv, Qb, Kb, Vt);
    attn_kernel<<<2048 / 8, 256, 0, stream>>>(Qb, Kb, Vt, ppT, AO);
    out_proj_kernel<<<4096 / 8, 256, 0, stream>>>(AO, woutT, bout, out);
}
